// HungarianMatcher_39084202394150
// MI455X (gfx1250) — matrix operand order repaired and checked by reading, not compiled, not run on hardware
//
#include <hip/hip_runtime.h>
#include <hip/hip_bf16.h>

// ---------------------------------------------------------------------------
// HungarianMatcher for MI455X (gfx1250, wave32, WMMA)
//   Phase 1: focal class-weight table W[q][c] (f16, padded C=91 -> 96)
//   Phase 2: cost_class = W x OneHot via v_wmma_f32_16x16x32_f16, fused with
//            L1 bbox cost and GIoU; writes transposed cost[g][q] (20MB, L2-resident)
//   Phase 3: Jonker-Volgenant LSA, single 1024-thread workgroup, parallel
//            column scans + LDS tree argmin; prefetch of active cost row.
// ---------------------------------------------------------------------------

typedef __attribute__((ext_vector_type(16))) _Float16 v16h;
typedef __attribute__((ext_vector_type(8)))  float    v8f;

#define QN 8192
#define GN 640
#define CN 91
#define CP 96            // padded class dim (multiple of 32 for WMMA K)
#define TPB 1024

#define ALPHA_F 0.25f

// ---- workspace layout (bytes) ----
#define OFF_W    0                                  // QN*CP f16  = 1,572,864
#define OFF_OH   (OFF_W + QN*CP*2)                  // GN*CP f16  =   122,880
#define OFF_COST (OFF_OH + GN*CP*2)                 // GN*QN f32  = 20,971,520
#define OFF_U    (OFF_COST + (size_t)GN*QN*4)       // (GN+16) f32
#define OFF_V    (OFF_U + (GN+16)*4)                // (QN+8) f32
#define OFF_WAY  (OFF_V + (QN+8)*4)                 // (QN+8) i32
#define OFF_P    (OFF_WAY + (QN+8)*4)               // (QN+8) i32

// ---------------------------------------------------------------------------
// Kernel 1: focal class weights (pos - neg), f16, zero-padded; zero one-hot.
__global__ __launch_bounds__(256) void hm_build_tables(
    const float* __restrict__ logits, _Float16* __restrict__ W,
    _Float16* __restrict__ OH)
{
    int idx = blockIdx.x * blockDim.x + threadIdx.x;
    if (idx < QN * CP) {
        int q = idx / CP, c = idx % CP;
        float val = 0.0f;
        if (c < CN) {
            float x = logits[q * CN + c];
            float p = 1.0f / (1.0f + __expf(-x));
            float pos = -ALPHA_F * (1.0f - p) * (1.0f - p) * __logf(p + 1e-6f);
            float neg = -(1.0f - ALPHA_F) * p * p * __logf(1.0f - p + 1e-6f);
            val = pos - neg;
        }
        W[idx] = (_Float16)val;
    }
    if (idx < GN * CP) OH[idx] = (_Float16)0.0f;
}

// Kernel 2: scatter ones into the one-hot (transposed: OH[g][c]).
__global__ void hm_set_onehot(const int* __restrict__ labels,
                              _Float16* __restrict__ OH)
{
    int g = blockIdx.x * blockDim.x + threadIdx.x;
    if (g < GN) {
        int l = labels[g];
        if (l < 0) l = 0; if (l >= CN) l = CN - 1;
        OH[g * CP + l] = (_Float16)1.0f;
    }
}

// ---------------------------------------------------------------------------
// Kernel 3: WMMA cost tiles. Each wave owns one 16(q) x 16(g) tile.
// cost_class via one-hot GEMM (3x v_wmma_f32_16x16x32_f16), then fused
// L1 + GIoU, written transposed: costT[g * QN + q].
__global__ __launch_bounds__(256) void hm_cost_wmma(
    const _Float16* __restrict__ W, const _Float16* __restrict__ OH,
    const float* __restrict__ pred_boxes, const float* __restrict__ gt_boxes,
    float* __restrict__ costT)
{
    const int lane = threadIdx.x & 31;
    const int wave = threadIdx.x >> 5;
    const int m0 = (blockIdx.x * 8 + wave) * 16;   // query tile base
    const int n0 = blockIdx.y * 16;                // gt tile base
    const int mr = lane & 15;                      // A row / B column / C col
    const int kh = lane >> 4;                      // K half selector

    v8f acc = {};
    const _Float16* Arow = W  + (size_t)(m0 + mr) * CP;
    const _Float16* Brow = OH + (size_t)(n0 + mr) * CP;
    for (int kb = 0; kb < CP; kb += 32) {
        v16h a, b;
#pragma unroll
        for (int v = 0; v < 8; ++v) {
            // A 16x32 f16 layout: VGPR v holds K = kh*8 + (v<4 ? 2v : 16+2(v-4)), K+1
            int ka = kb + kh * 8 + ((v < 4) ? 2 * v : 16 + 2 * (v - 4));
            a[2 * v]     = Arow[ka];
            a[2 * v + 1] = Arow[ka + 1];
            // B 32x16 f16 layout: lane = column n, VGPR v holds K = kh*8 + (v<4 ? 2v : 16+2(v-4)), K+1
            int kbv = kb + kh * 8 + ((v < 4) ? 2 * v : 16 + 2 * (v - 4));
            b[2 * v]     = Brow[kbv];
            b[2 * v + 1] = Brow[kbv + 1];
        }
        acc = __builtin_amdgcn_wmma_f32_16x16x32_f16(
            /*neg_a=*/false, a, /*neg_b=*/false, b,
            /*c_mod=*/(short)0, acc, /*reuse_a=*/false, /*reuse_b=*/false);
    }

    // gt box for this lane's column
    const int g = n0 + mr;
    const float gcx = gt_boxes[g * 4 + 0], gcy = gt_boxes[g * 4 + 1];
    const float gw  = gt_boxes[g * 4 + 2], gh  = gt_boxes[g * 4 + 3];
    const float gx1 = gcx - 0.5f * gw, gy1 = gcy - 0.5f * gh;
    const float gx2 = gcx + 0.5f * gw, gy2 = gcy + 0.5f * gh;
    const float area2 = (gx2 - gx1) * (gy2 - gy1);

#pragma unroll
    for (int r = 0; r < 8; ++r) {
        // C layout: VGPR r -> M = r + 8*kh, N = lane&15
        const int q = m0 + r + 8 * kh;
        const float pcx = pred_boxes[q * 4 + 0], pcy = pred_boxes[q * 4 + 1];
        const float pw  = pred_boxes[q * 4 + 2], ph  = pred_boxes[q * 4 + 3];
        const float l1 = fabsf(pcx - gcx) + fabsf(pcy - gcy) +
                         fabsf(pw - gw) + fabsf(ph - gh);
        const float px1 = pcx - 0.5f * pw, py1 = pcy - 0.5f * ph;
        const float px2 = pcx + 0.5f * pw, py2 = pcy + 0.5f * ph;
        const float area1 = (px2 - px1) * (py2 - py1);
        const float iw = fmaxf(fminf(px2, gx2) - fmaxf(px1, gx1), 0.0f);
        const float ih = fmaxf(fminf(py2, gy2) - fmaxf(py1, gy1), 0.0f);
        const float inter = iw * ih;
        const float uni = area1 + area2 - inter;
        const float iou = inter / uni;
        const float ew = fmaxf(px2, gx2) - fminf(px1, gx1);
        const float eh = fmaxf(py2, gy2) - fminf(py1, gy1);
        const float enc = fmaxf(ew, 0.0f) * fmaxf(eh, 0.0f);
        const float giou = iou - (enc - uni) / enc;
        costT[(size_t)g * QN + q] = acc[r] + l1 - giou;   // W_CLASS=W_BBOX=W_GIOU=1
    }
}

// ---------------------------------------------------------------------------
// Kernel 4: Jonker-Volgenant LSA on costT (GN rows x QN cols).
// Single workgroup, 1024 threads; minv + argmin reduction in LDS,
// potentials/matching in global ws (L2-resident).
__global__ __launch_bounds__(TPB) void hm_lsa(
    const float* __restrict__ costT,
    float* __restrict__ u, float* __restrict__ v,
    int* __restrict__ way, int* __restrict__ p,
    float* __restrict__ out)
{
    __shared__ float    s_minv[QN + 8];
    __shared__ float    s_redv[TPB];
    __shared__ int      s_redj[TPB];
    __shared__ unsigned s_used[QN / 32 + 2];
    __shared__ int      s_j0, s_i0;

    const int t = threadIdx.x;

    for (int j = t; j <= QN; j += TPB) { v[j] = 0.0f; p[j] = 0; way[j] = 0; }
    for (int r = t; r <= GN; r += TPB) u[r] = 0.0f;
    __syncthreads();

    for (int i = 1; i <= GN; ++i) {
        if (t == 0) { p[0] = i; s_j0 = 0; }
        for (int j = t; j <= QN; j += TPB) s_minv[j] = 1e30f;
        for (int w = t; w < QN / 32 + 2; w += TPB) s_used[w] = 0u;
        __syncthreads();

        for (;;) {
            if (t == 0) {
                int jj0 = s_j0;
                s_used[jj0 >> 5] |= (1u << (jj0 & 31));
                s_i0 = p[jj0];
            }
            __syncthreads();
            const int   j0  = s_j0;
            const int   i0  = s_i0;
            const float ui0 = u[i0];
            const float* crow = costT + (size_t)(i0 - 1) * QN;

            // scan free columns; keep thread-local (min, first-j)
            float lmin = 1e30f; int lj = -1;
#pragma unroll
            for (int c8 = 0; c8 < QN / TPB; ++c8) {
                const int j = 1 + t + c8 * TPB;
                if (!((s_used[j >> 5] >> (j & 31)) & 1u)) {
                    __builtin_prefetch(crow + (j - 1) + TPB, 0, 0);
                    const float cur = crow[j - 1] - ui0 - v[j];
                    float mv = s_minv[j];
                    if (cur < mv) { mv = cur; s_minv[j] = cur; way[j] = j0; }
                    if (mv < lmin) { lmin = mv; lj = j; }
                }
            }
            s_redv[t] = lmin; s_redj[t] = lj;
            __syncthreads();
            // lexicographic (value, j) tree argmin -> matches np.argmin ties
            for (int s = TPB / 2; s > 0; s >>= 1) {
                if (t < s) {
                    const float v1 = s_redv[t], v2 = s_redv[t + s];
                    const int   j1 = s_redj[t], j2 = s_redj[t + s];
                    const bool take = (j1 < 0) ||
                        (j2 >= 0 && (v2 < v1 || (v2 == v1 && j2 < j1)));
                    if (take) { s_redv[t] = v2; s_redj[t] = j2; }
                }
                __syncthreads();
            }
            const int   k     = s_redj[0];
            const float delta = s_redv[0];
            if (k < 0) break;   // defensive: no free column (cannot happen, QN >> GN)

            // potential updates: used -> u/v shift, free -> minv shift
            for (int j = t; j <= QN; j += TPB) {
                if ((s_used[j >> 5] >> (j & 31)) & 1u) {
                    u[p[j]] += delta;   // distinct rows per used column: race-free
                    v[j]    -= delta;
                } else if (j >= 1) {
                    s_minv[j] -= delta;
                }
            }
            if (t == 0) s_j0 = k;
            __threadfence_block();
            __syncthreads();
            if (p[k] == 0) break;
        }

        // augment along alternating path (serial, thread 0)
        if (t == 0) {
            int j0 = s_j0;
            while (j0) { const int j1 = way[j0]; p[j0] = p[j1]; j0 = j1; }
        }
        __threadfence_block();
        __syncthreads();
    }

    // emit matches in ascending query order (argsort of q_ind is identity)
    if (t == 0) {
        int cnt = 0;
        for (int j = 1; j <= QN && cnt < GN; ++j) {
            if (p[j]) {
                out[cnt]      = (float)(j - 1);   // src (query) index
                out[GN + cnt] = (float)(p[j] - 1); // tgt (gt) index
                ++cnt;
            }
        }
    }
}

// ---------------------------------------------------------------------------
extern "C" void kernel_launch(void* const* d_in, const int* in_sizes, int n_in,
                              void* d_out, int out_size, void* d_ws, size_t ws_size,
                              hipStream_t stream)
{
    const float* pred_boxes  = (const float*)d_in[0];  // (Q,4) f32
    const float* pred_logits = (const float*)d_in[1];  // (Q,91) f32
    const float* gt_boxes    = (const float*)d_in[2];  // (G,4) f32
    const int*   gt_labels   = (const int*)d_in[3];    // (G,) int

    char* ws = (char*)d_ws;
    _Float16* W     = (_Float16*)(ws + OFF_W);
    _Float16* OH    = (_Float16*)(ws + OFF_OH);
    float*    costT = (float*)(ws + OFF_COST);
    float*    u     = (float*)(ws + OFF_U);
    float*    v     = (float*)(ws + OFF_V);
    int*      way   = (int*)(ws + OFF_WAY);
    int*      p     = (int*)(ws + OFF_P);

    hm_build_tables<<<(QN * CP + 255) / 256, 256, 0, stream>>>(pred_logits, W, OH);
    hm_set_onehot<<<(GN + 63) / 64, 64, 0, stream>>>(gt_labels, OH);
    hm_cost_wmma<<<dim3(QN / 128, GN / 16), 256, 0, stream>>>(
        W, OH, pred_boxes, gt_boxes, costT);
    hm_lsa<<<1, TPB, 0, stream>>>(costT, u, v, way, p, (float*)d_out);
}
